// Block_CC_42125039239288
// MI455X (gfx1250) — matrix operand order repaired and checked by reading, not compiled, not run on hardware
//
#include <hip/hip_runtime.h>
#include <hip/hip_bf16.h>

typedef __bf16 bf16_t;
typedef __attribute__((ext_vector_type(16))) __bf16 v16bf;
typedef __attribute__((ext_vector_type(8)))  float  v8f;

constexpr int   kHeads = 4;
constexpr int   kKV    = 480;
constexpr int   kNPix  = 4096;   // 64 x 64
constexpr float kEps   = 1e-5f;

// ---------------- device helpers ----------------
static __device__ __forceinline__ bf16_t f2bf(float f) {
  union { float f; unsigned u; } in; in.f = f;
  unsigned r = (in.u + 0x7FFFu + ((in.u >> 16) & 1u)) >> 16;  // RNE truncate
  union { unsigned short s; bf16_t b; } out; out.s = (unsigned short)r;
  return out.b;
}
static __device__ __forceinline__ float wred_sum(float v) {
#pragma unroll
  for (int o = 16; o > 0; o >>= 1) v += __shfl_xor(v, o, 32);
  return v;
}
static __device__ __forceinline__ float wred_max(float v) {
#pragma unroll
  for (int o = 16; o > 0; o >>= 1) v = fmaxf(v, __shfl_xor(v, o, 32));
  return v;
}
static __device__ __forceinline__ float geluf(float x) {
  return 0.5f * x * (1.0f + erff(x * 0.70710678118654752f));
}

// ---------------- generic batched bf16 GEMM with WMMA ----------------
// C[M x N] (f32) = alpha * A(bf16, M x K row-major) * B(bf16)   (+ C if accum)
// bT==0: B is K x N row-major (ldb = row stride). bT==1: B is N x K row-major.
// Double-buffered LDS staging; interior tiles use CDNA5 async global->LDS DMA
// (ASYNCcnt), edge tiles fall back to guarded VGPR staging with zero-fill.
#define GEMM_BM 128
#define GEMM_BN 128
#define GEMM_BK 32
#define GEMM_LD 40   // padded LDS row stride (elements) -> 80B rows, 16B aligned

__global__ __launch_bounds__(256) void k_gemm(
    const bf16_t* __restrict__ A, const bf16_t* __restrict__ Bm, float* __restrict__ C,
    int M, int N, int K, int lda, int ldb, int ldc,
    long long sA, long long sB, long long sC,
    int bT, int accum, float alpha)
{
  __shared__ __align__(16) bf16_t As[2][GEMM_BM * GEMM_LD];
  __shared__ __align__(16) bf16_t Bs[2][GEMM_BN * GEMM_LD];
  A  += (long long)blockIdx.z * sA;
  Bm += (long long)blockIdx.z * sB;
  C  += (long long)blockIdx.z * sC;
  const int tM = blockIdx.y * GEMM_BM, tN = blockIdx.x * GEMM_BN;
  const int tid = threadIdx.x, lane = tid & 31, wave = tid >> 5;
  const int wM = (wave & 3) * 32, wN = (wave >> 2) * 64;
  const bf16_t zb = f2bf(0.0f);
  const int nk = (K + GEMM_BK - 1) / GEMM_BK;
  // async path requires 16B-aligned row starts and a fully interior tile edge
  const bool asyncA = ((lda & 7) == 0) && (tM + GEMM_BM <= M);
  const bool asyncB = bT && ((ldb & 7) == 0) && (tN + GEMM_BN <= N);

  auto stage = [&](int kt) {
    const int buf = kt & 1;
    const int k0 = kt * GEMM_BK;
    const bool intK = (k0 + GEMM_BK <= K);
    // ---- A tile: [GEMM_BM][GEMM_BK] row-major ----
    if (asyncA && intK) {
      const int r = tid >> 1, half = tid & 1;           // 256 thr = 128 rows x 2 x 32B
      const bf16_t* g = A + (long long)(tM + r) * lda + k0 + half * 16;
      unsigned la = (unsigned)(unsigned long long)&As[buf][r * GEMM_LD + half * 16];
      asm volatile("global_load_async_to_lds_b128 %0, %1, off"
                   :: "v"(la), "v"((unsigned long long)g) : "memory");
    } else {
      if (kt + 1 < nk) {
        int pr = tM + (tid >> 1);
        if (pr < M) __builtin_prefetch(A + (long long)pr * lda + (k0 + GEMM_BK), 0, 1);
      }
      for (int i = tid; i < GEMM_BM * GEMM_BK; i += 256) {
        int r = i >> 5, kk = i & (GEMM_BK - 1);
        int gr = tM + r, gk = k0 + kk;
        As[buf][r * GEMM_LD + kk] = (gr < M && gk < K) ? A[(long long)gr * lda + gk] : zb;
      }
    }
    // ---- B tile: staged N-major Bs[n][k] ----
    if (asyncB && intK) {
      const int n = tid >> 1, half = tid & 1;
      const bf16_t* g = Bm + (long long)(tN + n) * ldb + k0 + half * 16;
      unsigned lb = (unsigned)(unsigned long long)&Bs[buf][n * GEMM_LD + half * 16];
      asm volatile("global_load_async_to_lds_b128 %0, %1, off"
                   :: "v"(lb), "v"((unsigned long long)g) : "memory");
    } else {
      for (int i = tid; i < GEMM_BN * GEMM_BK; i += 256) {
        int n = i >> 5, kk = i & (GEMM_BK - 1);
        int gn = tN + n, gk = k0 + kk;
        bf16_t v = zb;
        if (gn < N && gk < K)
          v = bT ? Bm[(long long)gn * ldb + gk] : Bm[(long long)gk * ldb + gn];
        Bs[buf][n * GEMM_LD + kk] = v;
      }
    }
  };

  v8f acc[2][4];
#pragma unroll
  for (int i = 0; i < 2; ++i)
#pragma unroll
    for (int j = 0; j < 4; ++j)
#pragma unroll
      for (int e = 0; e < 8; ++e) acc[i][j][e] = 0.0f;

  stage(0);
  for (int kt = 0; kt < nk; ++kt) {
    // drain outstanding async tile DMA, then make it visible to all waves
    asm volatile("s_wait_asynccnt 0x0" ::: "memory");
    __syncthreads();
    if (kt + 1 < nk) stage(kt + 1);   // fill other buffer while computing this one

    const bf16_t* as = As[kt & 1];
    const bf16_t* bs = Bs[kt & 1];
    // fragments in contiguous k order (k = 16*(lane/16) + e), same for A and B
    const int m0  = lane & 15;
    const int kba = (lane < 16) ? 0 : 16;   // A: lanes16-31 hold K 16..31
    const int kbb = (lane < 16) ? 0 : 16;   // B: lanes16-31 hold K 16..31
    v16bf af[2], bfv[4];
#pragma unroll
    for (int i = 0; i < 2; ++i) {
      const bf16_t* p = &as[(wM + i * 16 + m0) * GEMM_LD];
#pragma unroll
      for (int e = 0; e < 16; ++e) af[i][e] = p[kba + e];
    }
#pragma unroll
    for (int j = 0; j < 4; ++j) {
      const bf16_t* p = &bs[(wN + j * 16 + m0) * GEMM_LD + kbb];
#pragma unroll
      for (int e = 0; e < 16; ++e) bfv[j][e] = p[e];
    }
#pragma unroll
    for (int i = 0; i < 2; ++i)
#pragma unroll
      for (int j = 0; j < 4; ++j)
        acc[i][j] = __builtin_amdgcn_wmma_f32_16x16x32_bf16(
            false, af[i], false, bfv[j], (short)0, acc[i][j], false, false);
    // no trailing barrier: each wave's ds reads are consumed by its WMMAs before
    // it reaches the next iteration's barrier; stage() writes the other buffer.
  }

  const int rowoff = (lane >> 4) * 8;
#pragma unroll
  for (int i = 0; i < 2; ++i)
#pragma unroll
    for (int j = 0; j < 4; ++j) {
      int col = tN + wN + j * 16 + (lane & 15);
      if (col >= N) continue;
#pragma unroll
      for (int v = 0; v < 8; ++v) {
        int row = tM + wM + i * 16 + rowoff + v;
        if (row < M) {
          long long o = (long long)row * ldc + col;
          float val = alpha * acc[i][j][v];
          C[o] = accum ? (C[o] + val) : val;
        }
      }
    }
}

// ---------------- channel LayerNorm for attention path ----------------
// wave (32 lanes) per pixel: per-scale LN -> cx_bf, concat-480 LN -> ea_bf
__global__ __launch_bounds__(128) void k_ln_attn(
    const float* __restrict__ e0, const float* __restrict__ e1,
    const float* __restrict__ e2, const float* __restrict__ e3,
    const float* __restrict__ w0, const float* __restrict__ bb0,
    const float* __restrict__ w1, const float* __restrict__ bb1,
    const float* __restrict__ w2, const float* __restrict__ bb2,
    const float* __restrict__ w3, const float* __restrict__ bb3,
    const float* __restrict__ wk, const float* __restrict__ bk,
    bf16_t* __restrict__ cx_bf, bf16_t* __restrict__ ea_bf, int Btot)
{
  const int lane = threadIdx.x & 31;
  const int wv = threadIdx.x >> 5;
  const int p = blockIdx.x * 4 + wv;
  if (p >= Btot * kNPix) return;
  const int b = p >> 12, hw = p & (kNPix - 1);
  const float* es[4] = {e0, e1, e2, e3};
  const float* ws[4] = {w0, w1, w2, w3};
  const float* bs[4] = {bb0, bb1, bb2, bb3};
  const int chs[4] = {32, 64, 128, 256};
  float xs[15];
  float s1t = 0.f, s2t = 0.f;
  int xi = 0;
  long long cxoff = 0;
  for (int s = 0; s < 4; ++s) {
    const int c = chs[s];
    float s1 = 0.f, s2 = 0.f;
    int k = 0;
    for (int ch = lane; ch < c; ch += 32, ++k) {
      float x = es[s][((long long)b * c + ch) * kNPix + hw];
      xs[xi + k] = x; s1 += x; s2 += x * x;
    }
    s1 = wred_sum(s1); s2 = wred_sum(s2);
    const float mu = s1 / c;
    const float inv = rsqrtf(s2 / c - mu * mu + kEps);
    k = 0;
    for (int ch = lane; ch < c; ch += 32, ++k)
      cx_bf[cxoff + ((long long)b * c + ch) * kNPix + hw] =
          f2bf((xs[xi + k] - mu) * inv * ws[s][ch] + bs[s][ch]);
    s1t += s1; s2t += s2; xi += c >> 5; cxoff += (long long)Btot * c * kNPix;
  }
  const float muA = s1t / kKV;
  const float invA = rsqrtf(s2t / kKV - muA * muA + kEps);
  int base = 0; xi = 0;
  for (int s = 0; s < 4; ++s) {
    const int c = chs[s];
    int k = 0;
    for (int ch = lane; ch < c; ch += 32, ++k)
      ea_bf[((long long)b * kKV + base + ch) * kNPix + hw] =
          f2bf((xs[xi + k] - muA) * invA * wk[base + ch] + bk[base + ch]);
    base += c; xi += c >> 5;
  }
}

// single-tensor channel LN (FFN pre-norm), f32 out
__global__ __launch_bounds__(128) void k_ln_single(
    const float* __restrict__ x, const float* __restrict__ w, const float* __restrict__ bb,
    float* __restrict__ y, int Btot, int C)
{
  const int lane = threadIdx.x & 31, wv = threadIdx.x >> 5;
  const int p = blockIdx.x * 4 + wv;
  if (p >= Btot * kNPix) return;
  const int b = p >> 12, hw = p & (kNPix - 1);
  float xs[8]; float s1 = 0.f, s2 = 0.f; int k = 0;
  for (int ch = lane; ch < C; ch += 32, ++k) {
    float v = x[((long long)b * C + ch) * kNPix + hw];
    xs[k] = v; s1 += v; s2 += v * v;
  }
  s1 = wred_sum(s1); s2 = wred_sum(s2);
  const float mu = s1 / C, inv = rsqrtf(s2 / C - mu * mu + kEps);
  k = 0;
  for (int ch = lane; ch < C; ch += 32, ++k)
    y[((long long)b * C + ch) * kNPix + hw] = (xs[k] - mu) * inv * w[ch] + bb[ch];
}

// row l2-norm over n columns, write bf16
__global__ __launch_bounds__(256) void k_l2norm_bf(const float* __restrict__ in,
                                                   bf16_t* __restrict__ out, int n)
{
  const float* p = in + (long long)blockIdx.x * n;
  bf16_t* o = out + (long long)blockIdx.x * n;
  const int lane = threadIdx.x & 31, wv = threadIdx.x >> 5;
  float s = 0.f;
  for (int i = threadIdx.x; i < n; i += 256) { float v = p[i]; s += v * v; }
  s = wred_sum(s);
  __shared__ float red[8];
  if (lane == 0) red[wv] = s;
  __syncthreads();
  if (threadIdx.x == 0) { float t = 0; for (int i = 0; i < 8; ++i) t += red[i]; red[0] = t; }
  __syncthreads();
  const float inv = 1.0f / fmaxf(sqrtf(red[0]), 1e-12f);
  for (int i = threadIdx.x; i < n; i += 256) o[i] = f2bf(p[i] * inv);
}

// f32 -> bf16 conversion
__global__ __launch_bounds__(256) void k_f32bf(const float* __restrict__ in,
                                               bf16_t* __restrict__ out, long long n)
{
  long long i = (long long)blockIdx.x * 256 + threadIdx.x;
  if (i < n) out[i] = f2bf(in[i]);
}

// grouped conv (cinPerG == coutPerG), optional bias + exact GELU
// input: in + b*inBstride + (inChOff + channel)*kNPix ; output packed [b][C][hw]
__global__ __launch_bounds__(256) void k_gconv(
    const float* __restrict__ in, long long inBstride, int inChOff,
    const float* __restrict__ w, const float* __restrict__ bias,
    float* __restrict__ out, int Btot, int C, int cig, int cog, int ks, int pad, int act)
{
  long long idx = (long long)blockIdx.x * 256 + threadIdx.x;
  long long total = (long long)Btot * C * kNPix;
  if (idx >= total) return;
  const int hw = (int)(idx & (kNPix - 1));
  const long long bc = idx >> 12;
  const int ch = (int)(bc % C), b = (int)(bc / C);
  const int x = hw & 63, y = hw >> 6;
  const int g = ch / cog;
  const int cib = g * cig;
  const float* wp = w + (long long)ch * cig * ks * ks;
  float a = bias ? bias[ch] : 0.f;
  for (int ci = 0; ci < cig; ++ci) {
    const float* ip = in + (long long)b * inBstride + (long long)(inChOff + cib + ci) * kNPix;
    for (int ky = 0; ky < ks; ++ky) {
      int yy = y + ky - pad;
      if ((unsigned)yy >= 64u) continue;
      for (int kx = 0; kx < ks; ++kx) {
        int xx = x + kx - pad;
        if ((unsigned)xx >= 64u) continue;
        a += ip[yy * 64 + xx] * wp[(ci * ks + ky) * ks + kx];
      }
    }
  }
  if (act) a = geluf(a);
  out[idx] = a;
}

// FFN partial conv: channels [0,dc) get full dc->dc 3x3 conv, rest copied
__global__ __launch_bounds__(256) void k_pcconv(
    const float* __restrict__ y, const float* __restrict__ w,
    float* __restrict__ xo, int Btot, int C, int dc)
{
  long long idx = (long long)blockIdx.x * 256 + threadIdx.x;
  long long total = (long long)Btot * C * kNPix;
  if (idx >= total) return;
  const int hw = (int)(idx & (kNPix - 1));
  const long long bc = idx >> 12;
  const int ch = (int)(bc % C), b = (int)(bc / C);
  if (ch >= dc) { xo[idx] = y[idx]; return; }
  const int x = hw & 63, y0 = hw >> 6;
  float a = 0.f;
  for (int ci = 0; ci < dc; ++ci) {
    const float* ip = y + ((long long)b * C + ci) * kNPix;
    const float* wp = w + ((long long)ch * dc + ci) * 9;
    for (int ky = 0; ky < 3; ++ky) {
      int yy = y0 + ky - 1;
      if ((unsigned)yy >= 64u) continue;
      for (int kx = 0; kx < 3; ++kx) {
        int xx = x + kx - 1;
        if ((unsigned)xx >= 64u) continue;
        a += ip[yy * 64 + xx] * wp[ky * 3 + kx];
      }
    }
  }
  xo[idx] = a;
}

// attention: whole-matrix LN over (c,480) then row softmax over 480, write bf16
__global__ __launch_bounds__(1024) void k_softmax(const float* __restrict__ attn,
                                                  bf16_t* __restrict__ out, int c, float scale)
{
  const long long zoff = (long long)blockIdx.x * c * kKV;
  const float* A = attn + zoff;
  bf16_t* O = out + zoff;
  const int tid = threadIdx.x, lane = tid & 31, wv = tid >> 5;
  __shared__ float r1[32], r2[32];
  const int total = c * kKV;
  float s1 = 0.f, s2 = 0.f;
  for (int i = tid; i < total; i += 1024) { float v = A[i] * scale; s1 += v; s2 += v * v; }
  s1 = wred_sum(s1); s2 = wred_sum(s2);
  if (lane == 0) { r1[wv] = s1; r2[wv] = s2; }
  __syncthreads();
  if (tid == 0) {
    float a = 0.f, b = 0.f;
    for (int i = 0; i < 32; ++i) { a += r1[i]; b += r2[i]; }
    r1[0] = a; r2[0] = b;
  }
  __syncthreads();
  const float mu = r1[0] / total;
  const float inv = rsqrtf(r2[0] / total - mu * mu + kEps);
  for (int r = wv; r < c; r += 32) {
    const float* row = A + (long long)r * kKV;
    float z[15]; float m = -3.0e38f;
#pragma unroll
    for (int k = 0; k < 15; ++k) {
      float v = (row[lane + 32 * k] * scale - mu) * inv;
      z[k] = v; m = fmaxf(m, v);
    }
    m = wred_max(m);
    float se = 0.f;
#pragma unroll
    for (int k = 0; k < 15; ++k) { z[k] = __expf(z[k] - m); se += z[k]; }
    se = wred_sum(se);
    const float rs = 1.0f / se;
#pragma unroll
    for (int k = 0; k < 15; ++k) O[(long long)r * kKV + lane + 32 * k] = f2bf(z[k] * rs);
  }
}

// elementwise helpers
__global__ __launch_bounds__(256) void k_add(const float* __restrict__ a,
                                             const float* __restrict__ b,
                                             float* __restrict__ o, long long n)
{
  long long i = (long long)blockIdx.x * 256 + threadIdx.x;
  if (i < n) o[i] = a[i] + b[i];
}
__global__ __launch_bounds__(256) void k_gelu_bias(float* __restrict__ t,
                                                   const float* __restrict__ bias,
                                                   int C, long long n)
{
  long long i = (long long)blockIdx.x * 256 + threadIdx.x;
  if (i < n) {
    int ch = (int)((i >> 12) % C);
    t[i] = geluf(t[i] + bias[ch]);
  }
}
__global__ __launch_bounds__(256) void k_mul(const float* __restrict__ a,
                                             const float* __restrict__ b,
                                             float* __restrict__ o, long long n)
{
  long long i = (long long)blockIdx.x * 256 + threadIdx.x;
  if (i < n) o[i] = a[i] * b[i];
}

// mean pool over hw (+ l2 bias), one block per (b,ch)
__global__ __launch_bounds__(256) void k_pool(const float* __restrict__ x,
                                              const float* __restrict__ l2b,
                                              float* __restrict__ pooled, int C)
{
  const int bc = blockIdx.x;
  const int ch = bc % C;
  const float* p = x + (long long)bc * kNPix;
  const int lane = threadIdx.x & 31, wv = threadIdx.x >> 5;
  float s = 0.f;
  for (int i = threadIdx.x; i < kNPix; i += 256) s += p[i];
  s = wred_sum(s);
  __shared__ float red[8];
  if (lane == 0) red[wv] = s;
  __syncthreads();
  if (threadIdx.x == 0) {
    float t = 0.f;
    for (int i = 0; i < 8; ++i) t += red[i];
    pooled[bc] = t / (float)kNPix + l2b[ch];
  }
}

// ECA: 1D conv (k=3, pad=1) over channel axis of pooled, sigmoid
__global__ __launch_bounds__(256) void k_eca(const float* __restrict__ pooled,
                                             const float* __restrict__ ew,
                                             float* __restrict__ s, int Btot, int C)
{
  int i = blockIdx.x * 256 + threadIdx.x;
  if (i >= Btot * C) return;
  int ch = i % C;
  float v = ew[1] * pooled[i];
  if (ch > 0) v += ew[0] * pooled[i - 1];
  if (ch < C - 1) v += ew[2] * pooled[i + 1];
  s[i] = 1.0f / (1.0f + __expf(-v));
}

// out = (xo + l2b) * s + cx2
__global__ __launch_bounds__(256) void k_finalize(const float* __restrict__ xo,
                                                  const float* __restrict__ l2b,
                                                  const float* __restrict__ s,
                                                  const float* __restrict__ cx2,
                                                  float* __restrict__ out, int C, long long n)
{
  long long i = (long long)blockIdx.x * 256 + threadIdx.x;
  if (i < n) {
    int bc = (int)(i >> 12);
    int ch = bc % C;
    out[i] = (xo[i] + l2b[ch]) * s[bc] + cx2[i];
  }
}

// ---------------- host orchestration ----------------
extern "C" void kernel_launch(void* const* d_in, const int* in_sizes, int n_in,
                              void* d_out, int out_size, void* d_ws, size_t ws_size,
                              hipStream_t stream)
{
  (void)in_sizes; (void)n_in; (void)out_size; (void)ws_size;
  const int B = 4;
  const int CHs[4] = {32, 64, 128, 256};
  const int HID[4] = {85, 170, 340, 680};   // int(c * 2.66)

  const float* emb[4];
  for (int i = 0; i < 4; ++i) emb[i] = (const float*)d_in[i];
  auto P  = [&](int i) { return (const float*)d_in[i]; };
  // per-scale param block: 4 + 17*i + j
  // j: 0 aln_w 1 aln_b 2 fln_w 3 fln_b 4 mh_w 5 q_w 6 po_w 7 pc_w 8 l1_w 9 l1_b
  //    10 dw3_w 11 dw3_b 12 dw5_w 13 dw5_b 14 l2_w 15 l2_b 16 eca_w
  auto PS = [&](int sc, int j) { return (const float*)d_in[4 + sc * 17 + j]; };
  const float* alnk_w = P(72); const float* alnk_b = P(73);
  const float* mhk_w  = P(74); const float* mhv_w  = P(75);
  const float* kcw    = P(76); const float* vcw    = P(77);

  // ---- workspace bump allocator ----
  char* wp_ = (char*)d_ws;
  auto alloc = [&](size_t bytes) -> void* {
    void* p = (void*)wp_;
    wp_ += (bytes + 255) & ~(size_t)255;
    return p;
  };

  const long long nEA = (long long)kKV * B * kNPix;       // 7,864,320
  const long long nKVfull = 1920LL * B * kNPix;           // 31,457,280
  const long long nCmax = 256LL * B * kNPix;              // 4,194,304

  bf16_t* ea_bf = (bf16_t*)alloc(nEA * 2);
  bf16_t* cx_bf = (bf16_t*)alloc(nEA * 2);
  bf16_t* wb_mhk = (bf16_t*)alloc(1920LL * 480 * 2);
  bf16_t* wb_mhv = (bf16_t*)alloc(1920LL * 480 * 2);
  bf16_t *wb_mh[4], *wb_po[4], *wb_l1[4], *wb_l2[4];
  for (int i = 0; i < 4; ++i) {
    long long c = CHs[i], h = HID[i];
    wb_mh[i] = (bf16_t*)alloc(4 * c * c * 2);
    wb_po[i] = (bf16_t*)alloc(c * c * 2);
    wb_l1[i] = (bf16_t*)alloc(2 * h * c * 2);
    wb_l2[i] = (bf16_t*)alloc(c * h * 2);
  }
  float*  buf1   = (float*)alloc(nKVfull * 4);
  float*  buf2   = (float*)alloc(nKVfull * 4);
  bf16_t* kk_bf  = (bf16_t*)alloc(nKVfull * 2);
  bf16_t* vv_bf  = (bf16_t*)alloc(nKVfull * 2);
  bf16_t* q_bf   = (bf16_t*)alloc(1024LL * B * kNPix * 2);
  float*  attn   = (float*)alloc((long long)B * kHeads * 256 * kKV * 4);
  bf16_t* attn_bf= (bf16_t*)alloc((long long)B * kHeads * 256 * kKV * 2);
  float*  attout = (float*)alloc(nCmax * 4);
  bf16_t* att_bf = (bf16_t*)alloc(nCmax * 2);
  float*  po_out = (float*)alloc(nCmax * 4);
  float*  cx2    = (float*)alloc(nEA * 4);
  float*  ybuf   = (float*)alloc(nCmax * 4);
  float*  xcat   = (float*)alloc(nCmax * 4);
  bf16_t* xcat_bf= (bf16_t*)alloc(nCmax * 2);
  float*  tbuf   = (float*)alloc(2LL * 680 * B * kNPix * 4);
  float*  abuf   = (float*)alloc(680LL * B * kNPix * 4);
  float*  b2buf  = (float*)alloc(680LL * B * kNPix * 4);
  bf16_t* prod_bf= (bf16_t*)alloc(680LL * B * kNPix * 2);
  float*  xo     = (float*)alloc(nCmax * 4);
  float*  pooled = (float*)alloc((size_t)B * 256 * 4);
  float*  svec   = (float*)alloc((size_t)B * 256 * 4);

  long long CXO[4];
  CXO[0] = 0;
  for (int i = 1; i < 4; ++i) CXO[i] = CXO[i - 1] + (long long)B * CHs[i - 1] * kNPix;

  auto conv_bf = [&](const float* src, bf16_t* dst, long long n) {
    k_f32bf<<<dim3((unsigned)((n + 255) / 256)), dim3(256), 0, stream>>>(src, dst, n);
  };
  auto gemm = [&](const bf16_t* A, const bf16_t* Bmat, float* C, int M, int N, int K,
                  int lda, int ldb, int ldc, long long sA, long long sB, long long sC,
                  int nb, int bT, int accum, float alpha) {
    dim3 g((N + GEMM_BN - 1) / GEMM_BN, (M + GEMM_BM - 1) / GEMM_BM, nb);
    k_gemm<<<g, dim3(256), 0, stream>>>(A, Bmat, C, M, N, K, lda, ldb, ldc,
                                        sA, sB, sC, bT, accum, alpha);
  };

  // ---- convert weights to bf16 ----
  conv_bf(mhk_w, wb_mhk, 1920LL * 480);
  conv_bf(mhv_w, wb_mhv, 1920LL * 480);
  for (int i = 0; i < 4; ++i) {
    long long c = CHs[i], h = HID[i];
    conv_bf(PS(i, 4),  wb_mh[i], 4 * c * c);
    conv_bf(PS(i, 6),  wb_po[i], c * c);
    conv_bf(PS(i, 8),  wb_l1[i], 2 * h * c);
    conv_bf(PS(i, 14), wb_l2[i], c * h);
  }

  // ---- channel LN (per scale + concat-480) ----
  k_ln_attn<<<dim3(B * kNPix / 4), dim3(128), 0, stream>>>(
      emb[0], emb[1], emb[2], emb[3],
      PS(0, 0), PS(0, 1), PS(1, 0), PS(1, 1), PS(2, 0), PS(2, 1), PS(3, 0), PS(3, 1),
      alnk_w, alnk_b, cx_bf, ea_bf, B);

  // ---- K path: pw -> depthwise 3x3 -> l2norm over n -> bf16 ----
  gemm(wb_mhk, ea_bf, buf1, 1920, kNPix, kKV, kKV, kNPix, kNPix,
       0LL, (long long)kKV * kNPix, 1920LL * kNPix, B, 0, 0, 1.0f);
  k_gconv<<<dim3((unsigned)((nKVfull + 255) / 256)), dim3(256), 0, stream>>>(
      buf1, 1920LL * kNPix, 0, kcw, nullptr, buf2, B, 1920, 1, 1, 3, 1, 0);
  k_l2norm_bf<<<dim3(1920 * B), dim3(256), 0, stream>>>(buf2, kk_bf, kNPix);

  // ---- V path: pw -> depthwise 3x3 -> bf16 ----
  gemm(wb_mhv, ea_bf, buf1, 1920, kNPix, kKV, kKV, kNPix, kNPix,
       0LL, (long long)kKV * kNPix, 1920LL * kNPix, B, 0, 0, 1.0f);
  k_gconv<<<dim3((unsigned)((nKVfull + 255) / 256)), dim3(256), 0, stream>>>(
      buf1, 1920LL * kNPix, 0, vcw, nullptr, buf2, B, 1920, 1, 1, 3, 1, 0);
  conv_bf(buf2, vv_bf, nKVfull);

  // ---- attention per scale ----
  const float scale = 1.0f / sqrtf((float)kKV);
  for (int i = 0; i < 4; ++i) {
    const int c = CHs[i];
    const long long nC = (long long)c * B * kNPix;
    const long long nQ = 4LL * c * B * kNPix;
    // q = l2norm(gconv(pw(cx_i, mh_w), q_w, 2-in/2-out groups))
    gemm(wb_mh[i], cx_bf + CXO[i], buf1, 4 * c, kNPix, c, c, kNPix, kNPix,
         0LL, (long long)c * kNPix, 4LL * c * kNPix, B, 0, 0, 1.0f);
    k_gconv<<<dim3((unsigned)((nQ + 255) / 256)), dim3(256), 0, stream>>>(
        buf1, 4LL * c * kNPix, 0, PS(i, 5), nullptr, buf2, B, 4 * c, 2, 2, 3, 1, 0);
    k_l2norm_bf<<<dim3(4 * c * B), dim3(256), 0, stream>>>(buf2, q_bf, kNPix);
    // attn[b,h] = q (c x n) * kk^T (n x 480)
    gemm(q_bf, kk_bf, attn, c, kKV, kNPix, kNPix, kNPix, kKV,
         (long long)c * kNPix, (long long)kKV * kNPix, (long long)c * kKV,
         B * kHeads, 1, 0, 1.0f);
    k_softmax<<<dim3(B * kHeads), dim3(1024), 0, stream>>>(attn, attn_bf, c, scale);
    // out = sum_h attn * vv / HEADS (accumulate heads sequentially)
    for (int h = 0; h < kHeads; ++h) {
      gemm(attn_bf + (long long)h * c * kKV, vv_bf + (long long)h * kKV * kNPix, attout,
           c, kNPix, kKV, kKV, kNPix, kNPix,
           (long long)kHeads * c * kKV, 1920LL * kNPix, (long long)c * kNPix,
           B, 0, (h > 0) ? 1 : 0, 0.25f);
    }
    conv_bf(attout, att_bf, nC);
    gemm(wb_po[i], att_bf, po_out, c, kNPix, c, c, kNPix, kNPix,
         0LL, (long long)c * kNPix, (long long)c * kNPix, B, 0, 0, 1.0f);
    k_add<<<dim3((unsigned)((nC + 255) / 256)), dim3(256), 0, stream>>>(
        emb[i], po_out, cx2 + CXO[i], nC);
  }

  // ---- FFN per scale ----
  for (int i = 0; i < 4; ++i) {
    const int c = CHs[i], h = HID[i], dc = c / 4;
    const float* cx2i = cx2 + CXO[i];
    const long long nC = (long long)c * B * kNPix;
    const long long nT = 2LL * h * B * kNPix;
    const long long nH = (long long)h * B * kNPix;
    k_ln_single<<<dim3(B * kNPix / 4), dim3(128), 0, stream>>>(cx2i, PS(i, 2), PS(i, 3), ybuf, B, c);
    k_pcconv<<<dim3((unsigned)((nC + 255) / 256)), dim3(256), 0, stream>>>(ybuf, PS(i, 7), xcat, B, c, dc);
    conv_bf(xcat, xcat_bf, nC);
    gemm(wb_l1[i], xcat_bf, tbuf, 2 * h, kNPix, c, c, kNPix, kNPix,
         0LL, (long long)c * kNPix, 2LL * h * kNPix, B, 0, 0, 1.0f);
    k_gelu_bias<<<dim3((unsigned)((nT + 255) / 256)), dim3(256), 0, stream>>>(tbuf, PS(i, 9), 2 * h, nT);
    k_gconv<<<dim3((unsigned)((nH + 255) / 256)), dim3(256), 0, stream>>>(
        tbuf, 2LL * h * kNPix, 0, PS(i, 10), PS(i, 11), abuf, B, h, 1, 1, 3, 1, 1);
    k_gconv<<<dim3((unsigned)((nH + 255) / 256)), dim3(256), 0, stream>>>(
        tbuf, 2LL * h * kNPix, h, PS(i, 12), PS(i, 13), b2buf, B, h, 1, 1, 5, 2, 1);
    k_mul<<<dim3((unsigned)((nH + 255) / 256)), dim3(256), 0, stream>>>(abuf, b2buf, abuf, nH);
    conv_bf(abuf, prod_bf, nH);
    gemm(wb_l2[i], prod_bf, xo, c, kNPix, h, h, kNPix, kNPix,
         0LL, (long long)h * kNPix, (long long)c * kNPix, B, 0, 0, 1.0f);
    k_pool<<<dim3(B * c), dim3(256), 0, stream>>>(xo, PS(i, 15), pooled, c);
    k_eca<<<dim3((unsigned)((B * c + 255) / 256)), dim3(256), 0, stream>>>(pooled, PS(i, 16), svec, B, c);
    float* outp = (float*)d_out + CXO[i];
    k_finalize<<<dim3((unsigned)((nC + 255) / 256)), dim3(256), 0, stream>>>(
        xo, PS(i, 15), svec, cx2i, outp, c, nC);
  }
}
